// Regress_graph_gc_128849019553
// MI455X (gfx1250) — compile-verified, matrix operand order checked by reading, not run on hardware
//
#include <hip/hip_runtime.h>
#include <hip/hip_bf16.h>
#include <math.h>

typedef __attribute__((ext_vector_type(16))) __bf16 v16bf;
typedef __attribute__((ext_vector_type(8)))  __bf16 v8bf;
typedef __attribute__((ext_vector_type(8)))  float  v8f;

#define HID 128

// ---------------------------------------------------------------------------
// Degree / normalization
// ---------------------------------------------------------------------------
__global__ void gcn_init_deg(float* deg, int nNodes) {
    int i = blockIdx.x * blockDim.x + threadIdx.x;
    if (i < nNodes) deg[i] = 1.0f;                 // self-loop contribution
}

__global__ void gcn_accum_deg(const int* __restrict__ col, float* deg, int nEdges) {
    int e = blockIdx.x * blockDim.x + threadIdx.x;
    if (e < nEdges) unsafeAtomicAdd(&deg[col[e]], 1.0f);
}

__global__ void gcn_deg_to_dinv(float* deg, int nNodes) {
    int i = blockIdx.x * blockDim.x + threadIdx.x;
    if (i < nNodes) deg[i] = rsqrtf(fmaxf(deg[i], 1.0f));   // deg >= 1 always
}

// ---------------------------------------------------------------------------
// Operand preparation: f32 -> bf16 copies for the WMMA GEMM.
//   Hb[i]      = (bf16) H[i]                (N x 128, row major)
//   Wt[n][k]   = (bf16) W[k][n]             (transposed so B-fragment loads
//                                            are two contiguous 16B chunks)
// ---------------------------------------------------------------------------
__global__ void gcn_cvt_bf16(const float* __restrict__ src,
                             __bf16* __restrict__ dst, int total) {
    int i = blockIdx.x * blockDim.x + threadIdx.x;
    if (i < total) dst[i] = (__bf16)src[i];
}

__global__ void gcn_prep_wt(const float* __restrict__ W, __bf16* __restrict__ Wt) {
    int i = blockIdx.x * blockDim.x + threadIdx.x;   // 16384 threads
    if (i >= HID * HID) return;
    int k = i >> 7, n = i & (HID - 1);
    Wt[(size_t)n * HID + k] = (__bf16)W[i];
}

// ---------------------------------------------------------------------------
// GEMM: XW = H (N x 128) * W (128 x 128) via bf16 WMMA, f32 accumulate.
// Block = 256 thr = 8 waves; block covers rows [16*bx, 16*bx+16), wave w
// covers cols [16*w, 16*w+16). K = 128 -> 4 unrolled v_wmma_f32_16x16x32_bf16.
// Per the ISA 16-bit A/B layout, lane (mn, kg) holds K ranges
// [kg*8, kg*8+8) and [16+kg*8, 16+kg*8+8) -> two aligned 16B loads/operand.
// ---------------------------------------------------------------------------
__global__ __launch_bounds__(256) void gcn_gemm_wmma(
    const __bf16* __restrict__ Hb, const __bf16* __restrict__ Wt,
    float* __restrict__ XW, int nNodes) {
    const int wave    = threadIdx.x >> 5;
    const int lane    = threadIdx.x & 31;
    const int rowBase = blockIdx.x * 16;
    const int colBase = wave * 16;
    const int mn      = lane & 15;   // M for A-frag, N for B-frag
    const int kg      = lane >> 4;   // K-octet select

    const int arow = min(rowBase + mn, nNodes - 1);
    const __bf16* aPtr = Hb + (size_t)arow * HID + kg * 8;
    const __bf16* bPtr = Wt + (size_t)(colBase + mn) * HID + kg * 8;

    v8f acc = {};
#pragma unroll
    for (int kc = 0; kc < HID; kc += 32) {
        v8bf a0 = *(const v8bf*)(aPtr + kc);
        v8bf a1 = *(const v8bf*)(aPtr + kc + 16);
        v8bf b0 = *(const v8bf*)(bPtr + kc);
        v8bf b1 = *(const v8bf*)(bPtr + kc + 16);
        v16bf a = __builtin_shufflevector(a0, a1, 0, 1, 2, 3, 4, 5, 6, 7,
                                          8, 9, 10, 11, 12, 13, 14, 15);
        v16bf b = __builtin_shufflevector(b0, b1, 0, 1, 2, 3, 4, 5, 6, 7,
                                          8, 9, 10, 11, 12, 13, 14, 15);
        acc = __builtin_amdgcn_wmma_f32_16x16x32_bf16(
            false, a, false, b, (short)0, acc, false, false);
    }

    // C/D layout: VGPR r -> M = r (lanes 0-15) or r+8 (lanes 16-31), N = lane%16
    const int mOff = (lane >> 4) * 8;
    if (rowBase + 16 <= nNodes) {            // uniform fast path (always taken
        float* orow = XW + (size_t)(rowBase + mOff) * HID + colBase + mn;
#pragma unroll                               //  when nNodes % 16 == 0)
        for (int r = 0; r < 8; ++r) orow[(size_t)r * HID] = acc[r];
    } else {
#pragma unroll
        for (int r = 0; r < 8; ++r) {
            const int mm = rowBase + mOff + r;
            if (mm < nNodes) XW[(size_t)mm * HID + colBase + mn] = acc[r];
        }
    }
}

// ---------------------------------------------------------------------------
// out[i][f] = dinv[i]^2 * xw[i][f]   (self-loop term; initializes out)
// ---------------------------------------------------------------------------
__global__ void gcn_self_loop_init(const float* __restrict__ xw,
                                   const float* __restrict__ dinv,
                                   float* __restrict__ outb, int nNodes) {
    int i = blockIdx.x * blockDim.x + threadIdx.x;
    if (i >= nNodes * HID) return;
    float d = dinv[i >> 7];
    outb[i] = d * d * xw[i];
}

// ---------------------------------------------------------------------------
// One wave per edge: out[col] += dinv[row]*dinv[col] * xw[row]  (128 floats)
// ---------------------------------------------------------------------------
__global__ __launch_bounds__(256) void gcn_edge_scatter(
    const float* __restrict__ xw, const int* __restrict__ row,
    const int* __restrict__ col, const float* __restrict__ dinv,
    float* __restrict__ outb, int nEdges) {
    int e    = (int)((blockIdx.x * blockDim.x + threadIdx.x) >> 5);
    int lane = threadIdx.x & 31;
    if (e >= nEdges) return;
    int r = row[e], c = col[e];
    float wgt = dinv[r] * dinv[c];
    const float4 v = *(const float4*)(xw + (size_t)r * HID + lane * 4);
    float* dst = outb + (size_t)c * HID + lane * 4;
    unsafeAtomicAdd(&dst[0], wgt * v.x);
    unsafeAtomicAdd(&dst[1], wgt * v.y);
    unsafeAtomicAdd(&dst[2], wgt * v.z);
    unsafeAtomicAdd(&dst[3], wgt * v.w);
}

// ---------------------------------------------------------------------------
// h = elu(out + b)  in place; also emit bf16 copy for the next layer's GEMM.
// ---------------------------------------------------------------------------
__global__ void gcn_bias_elu(float* __restrict__ outb,
                             const float* __restrict__ bias,
                             __bf16* __restrict__ Hb, int total) {
    int i = blockIdx.x * blockDim.x + threadIdx.x;
    if (i >= total) return;
    float v = outb[i] + bias[i & (HID - 1)];
    v = v > 0.0f ? v : (__expf(v) - 1.0f);
    outb[i] = v;
    Hb[i]   = (__bf16)v;
}

// ---------------------------------------------------------------------------
// Pooling
// ---------------------------------------------------------------------------
__global__ void gcn_zero_pool(float* pooled, float* counts, int nGraphs) {
    int i = blockIdx.x * blockDim.x + threadIdx.x;
    if (i < nGraphs * HID) pooled[i] = 0.0f;
    if (i < nGraphs) counts[i] = 0.0f;
}

__global__ __launch_bounds__(256) void gcn_pool_accum(
    const float* __restrict__ h, const int* __restrict__ batch,
    float* pooled, float* counts, int nNodes) {
    int node = (int)((blockIdx.x * blockDim.x + threadIdx.x) >> 5);
    int lane = threadIdx.x & 31;
    if (node >= nNodes) return;
    int g = batch[node];
    const float4 v = *(const float4*)(h + (size_t)node * HID + lane * 4);
    float* pr = pooled + (size_t)g * HID + lane * 4;
    unsafeAtomicAdd(&pr[0], v.x);
    unsafeAtomicAdd(&pr[1], v.y);
    unsafeAtomicAdd(&pr[2], v.z);
    unsafeAtomicAdd(&pr[3], v.w);
    if (lane == 0) unsafeAtomicAdd(&counts[g], 1.0f);
}

__global__ __launch_bounds__(128) void gcn_final(
    const float* __restrict__ pooled, const float* __restrict__ counts,
    const float* __restrict__ Wout, const float* __restrict__ bout,
    float* __restrict__ out) {
    __shared__ float red[HID];
    int g = blockIdx.x, t = threadIdx.x;
    float inv = 1.0f / fmaxf(counts[g], 1.0f);
    red[t] = pooled[(size_t)g * HID + t] * inv * Wout[t];
    __syncthreads();
    for (int s = 64; s > 0; s >>= 1) {
        if (t < s) red[t] += red[t + s];
        __syncthreads();
    }
    if (t == 0) out[g] = red[0] + bout[0];
}

// ---------------------------------------------------------------------------
extern "C" void kernel_launch(void* const* d_in, const int* in_sizes, int n_in,
                              void* d_out, int out_size, void* d_ws, size_t ws_size,
                              hipStream_t stream) {
    const float* x    = (const float*)d_in[0];
    const float* W0   = (const float*)d_in[1];
    const float* b0   = (const float*)d_in[2];
    const float* W1   = (const float*)d_in[3];
    const float* b1   = (const float*)d_in[4];
    const float* W2   = (const float*)d_in[5];
    const float* b2   = (const float*)d_in[6];
    const float* Wout = (const float*)d_in[7];
    const float* bout = (const float*)d_in[8];
    const int*   eidx = (const int*)d_in[9];
    const int*   batch= (const int*)d_in[10];

    const int nNodes  = in_sizes[0] / HID;     // 50000
    const int nEdges  = in_sizes[9] / 2;       // 800000
    const int nGraphs = out_size;              // 500
    const int* erow = eidx;
    const int* ecol = eidx + nEdges;

    // Workspace carve-up (all chunks 16B-aligned sizes)
    char* ws = (char*)d_ws;
    const size_t featBytes = (size_t)nNodes * HID * sizeof(float);    // 25.6 MB
    const size_t featBytesH = (size_t)nNodes * HID * sizeof(__bf16);  // 12.8 MB
    float*  xw     = (float*)ws;     ws += featBytes;
    float*  outb   = (float*)ws;     ws += featBytes;
    __bf16* Hb     = (__bf16*)ws;    ws += featBytesH;
    __bf16* Wt     = (__bf16*)ws;    ws += (size_t)HID * HID * sizeof(__bf16);
    float*  dinv   = (float*)ws;     ws += (size_t)nNodes * sizeof(float);
    float*  pooled = (float*)ws;     ws += (size_t)nGraphs * HID * sizeof(float);
    float*  counts = (float*)ws;

    float* outp = (float*)d_out;

    // --- normalization ---
    gcn_init_deg<<<(nNodes + 255) / 256, 256, 0, stream>>>(dinv, nNodes);
    gcn_accum_deg<<<(nEdges + 255) / 256, 256, 0, stream>>>(ecol, dinv, nEdges);
    gcn_deg_to_dinv<<<(nNodes + 255) / 256, 256, 0, stream>>>(dinv, nNodes);

    // --- 3 GCN layers ---
    const float* Wl[3] = {W0, W1, W2};
    const float* bl[3] = {b0, b1, b2};
    const int totalFeat = nNodes * HID;

    // first layer input in bf16
    gcn_cvt_bf16<<<(totalFeat + 255) / 256, 256, 0, stream>>>(x, Hb, totalFeat);

    for (int l = 0; l < 3; ++l) {
        gcn_prep_wt<<<(HID * HID + 255) / 256, 256, 0, stream>>>(Wl[l], Wt);
        gcn_gemm_wmma<<<(nNodes + 15) / 16, 256, 0, stream>>>(Hb, Wt, xw, nNodes);
        gcn_self_loop_init<<<(totalFeat + 255) / 256, 256, 0, stream>>>(xw, dinv, outb, nNodes);
        gcn_edge_scatter<<<(nEdges + 7) / 8, 256, 0, stream>>>(xw, erow, ecol, dinv, outb, nEdges);
        gcn_bias_elu<<<(totalFeat + 255) / 256, 256, 0, stream>>>(outb, bl[l], Hb, totalFeat);
    }

    // --- global mean pool + linear head ---
    gcn_zero_pool<<<(nGraphs * HID + 255) / 256, 256, 0, stream>>>(pooled, counts, nGraphs);
    gcn_pool_accum<<<(nNodes + 7) / 8, 256, 0, stream>>>(outb, batch, pooled, counts, nNodes);
    gcn_final<<<nGraphs, 128, 0, stream>>>(pooled, counts, Wout, bout, outp);
}
